// SlidingWindowAttention_2731599200632
// MI455X (gfx1250) — compile-verified, matrix operand order checked by reading, not run on hardware
//
#include <hip/hip_runtime.h>

#define D_MODEL   1024
#define NUM_HEADS 16
#define HEAD_DIM  64
#define WINDOW    128
#define SEQ       2048
#define SCALE_F   0.125f   // 64^-0.5

typedef __attribute__((ext_vector_type(16))) _Float16 v16h;
typedef __attribute__((ext_vector_type(8)))  _Float16 v8h;
typedef __attribute__((ext_vector_type(4)))  _Float16 v4h;
typedef __attribute__((ext_vector_type(8)))  float    v8f;
typedef __attribute__((ext_vector_type(4)))  float    v4f;

#define WMMA_F16(A_, B_, C_) \
  __builtin_amdgcn_wmma_f32_16x16x32_f16(false, (A_), false, (B_), (short)0, (C_), false, false)

// Build a 16x32 f16 WMMA fragment (A or B) from a row whose 32 K-values are
// contiguous in memory. Per ISA 16-bit A layout: lanes 0-15 hold K {0..7,16..23},
// lanes 16-31 hold K {8..15,24..31} -> two 16-byte contiguous loads per lane.
static __device__ __forceinline__ v16h load_frag(const _Float16* row, int lane) {
  const int kb = (lane < 16) ? 0 : 8;
  v8h lo = *(const v8h*)(row + kb);
  v8h hi = *(const v8h*)(row + kb + 16);
  v16h r;
#pragma unroll
  for (int i = 0; i < 8; ++i) { r[i] = lo[i]; r[i + 8] = hi[i]; }
  return r;
}

__global__ void cvt_f32_f16(const float* __restrict__ in, _Float16* __restrict__ out, int n4) {
  for (int i = blockIdx.x * blockDim.x + threadIdx.x; i < n4; i += gridDim.x * blockDim.x) {
    v4f v = ((const v4f*)in)[i];
    v4h o;
#pragma unroll
    for (int e = 0; e < 4; ++e) o[e] = (_Float16)v[e];
    ((v4h*)out)[i] = o;
  }
}

// Vh: [B, S, H*64] -> Vt: [B*H, 64, S], LDS-tiled so both sides coalesce.
// One block per (bh, 64-row s-tile).
__global__ __launch_bounds__(256) void transpose_v(const _Float16* __restrict__ Vh,
                                                   _Float16* __restrict__ Vt) {
  __shared__ _Float16 tile[64][66];  // +2 f16 pad -> odd dword stride
  const int bh = blockIdx.x >> 5;          // SEQ/64 = 32 s-tiles per (b,h)
  const int s0 = (blockIdx.x & 31) << 6;
  const int b  = bh >> 4, h = bh & 15;
  const int t  = threadIdx.x;
#pragma unroll
  for (int it = 0; it < 16; ++it) {
    const int lin = it * 256 + t;
    const int si = lin >> 6, dj = lin & 63;
    tile[si][dj] = Vh[(size_t)(b * SEQ + s0 + si) * D_MODEL + h * HEAD_DIM + dj];
  }
  __syncthreads();
#pragma unroll
  for (int it = 0; it < 16; ++it) {
    const int lin = it * 256 + t;
    const int d = lin >> 6, sj = lin & 63;
    Vt[((size_t)bh * HEAD_DIM + d) * SEQ + s0 + sj] = tile[sj][d];
  }
}

// C[M,N] = A[M,K] * W[N,K]^T ; one wave computes a 32x64 tile:
// two 16-row A fragments share each B fragment -> 8 WMMA per 32-wide K step.
template <bool F16OUT>
__global__ __launch_bounds__(256) void gemm_xwt(const _Float16* __restrict__ A,
                                                const _Float16* __restrict__ W,
                                                void* __restrict__ out,
                                                int M, int N, int K) {
  const int lane = threadIdx.x & 31;
  const int wid  = (blockIdx.x * blockDim.x + threadIdx.x) >> 5;
  const int tilesN = N >> 6;
  const int m0 = (wid / tilesN) << 5;
  const int n0 = (wid % tilesN) << 6;
  if (m0 >= M) return;
  const int col = lane & 15;
  const int lh  = lane >> 4;

  v8f acc0[4] = {v8f{}, v8f{}, v8f{}, v8f{}};
  v8f acc1[4] = {v8f{}, v8f{}, v8f{}, v8f{}};
  const _Float16* arow0 = A + (size_t)(m0 + col) * K;
  const _Float16* arow1 = A + (size_t)(m0 + 16 + col) * K;

  for (int k0 = 0; k0 < K; k0 += 32) {
    v16h af0 = load_frag(arow0 + k0, lane);
    v16h af1 = load_frag(arow1 + k0, lane);
#pragma unroll
    for (int t = 0; t < 4; ++t) {
      v16h bf = load_frag(W + (size_t)(n0 + t * 16 + col) * K + k0, lane);
      acc0[t] = WMMA_F16(af0, bf, acc0[t]);
      acc1[t] = WMMA_F16(af1, bf, acc1[t]);
    }
  }
#pragma unroll
  for (int t = 0; t < 4; ++t)
#pragma unroll
    for (int i = 0; i < 8; ++i) {
      size_t idx0 = (size_t)(m0 + i + lh * 8) * N + (n0 + t * 16 + col);
      size_t idx1 = (size_t)(m0 + 16 + i + lh * 8) * N + (n0 + t * 16 + col);
      if (F16OUT) {
        ((_Float16*)out)[idx0] = (_Float16)acc0[t][i];
        ((_Float16*)out)[idx1] = (_Float16)acc1[t][i];
      } else {
        ((float*)out)[idx0] = acc0[t][i];
        ((float*)out)[idx1] = acc1[t][i];
      }
    }
}

// Flash-style banded attention. One wave = one 16-query tile of one (b,h).
// Scores bounce through LDS once (D-layout -> per-lane A-fragment pattern).
// Interior key-tiles (fully inside the band for all 16 queries) take a
// wave-uniform unmasked fast path; only edge tiles pay for mask compares.
__global__ __launch_bounds__(256) void attn_band(const _Float16* __restrict__ Qh,
                                                 const _Float16* __restrict__ Kh,
                                                 const _Float16* __restrict__ Vt,
                                                 _Float16* __restrict__ Oh, int Bsz) {
  __shared__ __align__(16) float ldsS[8][16 * 32];  // per-wave f32 score scratch
  const int lane  = threadIdx.x & 31;
  const int wslot = threadIdx.x >> 5;
  const int wid   = (blockIdx.x * blockDim.x + threadIdx.x) >> 5;
  const int qtiles = SEQ / 16;
  const int qt = wid % qtiles;
  const int bh = wid / qtiles;
  if (bh >= Bsz * NUM_HEADS) return;
  const int b = bh / NUM_HEADS;
  const int h = bh - b * NUM_HEADS;
  const int q0 = qt * 16;
  const int col = lane & 15;            // D-layout column / fragment row
  const int lh  = lane >> 4;
  const int kb  = (lane < 16) ? 0 : 8;  // this lane's K-half inside a fragment
  const int q   = q0 + col;             // query row this lane owns for softmax
  float* Sw = &ldsS[wslot][0];

  const _Float16* Qbase = Qh + (size_t)(b * SEQ + q0) * D_MODEL + h * HEAD_DIM;
  v16h qa0 = load_frag(Qbase + (size_t)col * D_MODEL, lane);       // dk 0..31
  v16h qa1 = load_frag(Qbase + (size_t)col * D_MODEL + 32, lane);  // dk 32..63
#pragma unroll
  for (int e = 0; e < 16; ++e) {  // fold softmax scale into Q (exact: x0.125)
    qa0[e] *= (_Float16)SCALE_F;
    qa1[e] *= (_Float16)SCALE_F;
  }

  float m_row = -1e30f, l_row = 0.f;
  v8f o[4] = {v8f{}, v8f{}, v8f{}, v8f{}};

  const int jlo = (q0 > WINDOW) ? ((q0 - WINDOW) & ~31) : 0;
  int jhi = q0 + 15 + WINDOW; if (jhi > SEQ - 1) jhi = SEQ - 1;

  for (int j0 = jlo; j0 <= jhi; j0 += 32) {
    // ---- scores for 32 keys as two 16x16 WMMA tiles (pre-scaled) ----
    const _Float16* Kb0 = Kh + (size_t)(b * SEQ + j0) * D_MODEL + h * HEAD_DIM
                             + (size_t)col * D_MODEL;
    v8f s0 = {}, s1 = {};
    s0 = WMMA_F16(qa0, load_frag(Kb0, lane), s0);
    s0 = WMMA_F16(qa1, load_frag(Kb0 + 32, lane), s0);
    const _Float16* Kb1 = Kb0 + (size_t)16 * D_MODEL;
    s1 = WMMA_F16(qa0, load_frag(Kb1, lane), s1);
    s1 = WMMA_F16(qa1, load_frag(Kb1 + 32, lane), s1);

    // ---- D-layout -> LDS (raw f32 scores) ----
#pragma unroll
    for (int i = 0; i < 8; ++i) {
      Sw[(i + lh * 8) * 32 + col]      = s0[i];
      Sw[(i + lh * 8) * 32 + 16 + col] = s1[i];
    }
    asm volatile("s_wait_dscnt 0" ::: "memory");  // cross-lane LDS visibility

    // ---- each lane reads its own P-fragment's 16 scores: row=col, K-half=kb ----
    float sv[16];
    const float* Srow = Sw + col * 32 + kb;
#pragma unroll
    for (int e = 0; e < 8; ++e) { sv[e] = Srow[e]; sv[e + 8] = Srow[e + 16]; }

    // tile fully inside band for all 16 query rows?  (wave-uniform -> SALU branch)
    const bool full = (j0 >= q0 - (WINDOW - 15)) && (j0 <= q0 + (WINDOW - 31));

    float mnew, alpha, sloc;
    v16h pf;
    if (full) {
      float mloc = sv[0];
#pragma unroll
      for (int e = 1; e < 16; ++e) mloc = fmaxf(mloc, sv[e]);
      const float rowmax = fmaxf(mloc, __shfl_xor(mloc, 16));
      mnew  = fmaxf(m_row, rowmax);
      alpha = __expf(m_row - mnew);
      sloc  = 0.f;
#pragma unroll
      for (int e = 0; e < 16; ++e) {
        const float pv = __expf(sv[e] - mnew);
        pf[e] = (_Float16)pv;
        sloc += pv;
      }
    } else {
      bool vm[16];
      float mloc = -1e30f;
#pragma unroll
      for (int e = 0; e < 16; ++e) {
        const int j = j0 + kb + ((e < 8) ? e : e + 8);
        vm[e] = (j >= q - WINDOW) && (j <= q + WINDOW);
        sv[e] = vm[e] ? sv[e] : -1e30f;
        mloc  = fmaxf(mloc, sv[e]);
      }
      const float rowmax = fmaxf(mloc, __shfl_xor(mloc, 16));
      mnew  = fmaxf(m_row, rowmax);
      alpha = __expf(m_row - mnew);
      sloc  = 0.f;
#pragma unroll
      for (int e = 0; e < 16; ++e) {
        const float pv = vm[e] ? __expf(sv[e] - mnew) : 0.f;
        pf[e] = (_Float16)pv;
        sloc += pv;
      }
    }
    const float rowsum = sloc + __shfl_xor(sloc, 16);
    l_row = l_row * alpha + rowsum;
    m_row = mnew;

    // ---- rescale O (D-layout): broadcast per-row alpha from owning lane ----
#pragma unroll
    for (int i = 0; i < 8; ++i) {
      const float ai = __shfl(alpha, i + lh * 8);
      o[0][i] *= ai; o[1][i] *= ai; o[2][i] *= ai; o[3][i] *= ai;
    }

    // ---- O += P * V ----
    const _Float16* Vb = Vt + (size_t)bh * HEAD_DIM * SEQ + j0;
#pragma unroll
    for (int t = 0; t < 4; ++t) {
      v16h vf = load_frag(Vb + (size_t)(t * 16 + col) * SEQ, lane);
      o[t] = WMMA_F16(pf, vf, o[t]);
    }
  }

  const float invl = 1.f / l_row;
  _Float16* Ob = Oh + (size_t)(b * SEQ + q0) * D_MODEL + h * HEAD_DIM;
#pragma unroll
  for (int i = 0; i < 8; ++i) {
    const float li = __shfl(invl, i + lh * 8);
#pragma unroll
    for (int t = 0; t < 4; ++t)
      Ob[(size_t)(i + lh * 8) * D_MODEL + t * 16 + col] = (_Float16)(o[t][i] * li);
  }
}

extern "C" void kernel_launch(void* const* d_in, const int* in_sizes, int n_in,
                              void* d_out, int out_size, void* d_ws, size_t ws_size,
                              hipStream_t stream) {
  (void)n_in; (void)out_size; (void)ws_size;
  const float* x  = (const float*)d_in[0];
  const float* Wq = (const float*)d_in[1];
  const float* Wk = (const float*)d_in[2];
  const float* Wv = (const float*)d_in[3];
  const float* Wo = (const float*)d_in[4];

  const int BS  = in_sizes[0] / D_MODEL;   // B*S
  const int Bsz = BS / SEQ;
  const size_t nx = (size_t)BS * D_MODEL;      // elements in x/Q/K/V/Vt/attn
  const size_t nw = (size_t)D_MODEL * D_MODEL; // elements per weight

  _Float16* p   = (_Float16*)d_ws;
  _Float16* xh  = p; p += nx;
  _Float16* Wqh = p; p += nw;
  _Float16* Wkh = p; p += nw;
  _Float16* Wvh = p; p += nw;
  _Float16* Woh = p; p += nw;
  _Float16* Qh  = p; p += nx;
  _Float16* Khd = p; p += nx;
  _Float16* Vh  = p; p += nx;
  _Float16* Vtp = p; p += nx;   // [B*H, 64, S] == BS*D elements
  _Float16* Ah  = p; p += nx;

  const int T = 256;
  cvt_f32_f16<<<(int)((nx / 4 + T - 1) / T), T, 0, stream>>>(x,  xh,  (int)(nx / 4));
  cvt_f32_f16<<<(int)((nw / 4 + T - 1) / T), T, 0, stream>>>(Wq, Wqh, (int)(nw / 4));
  cvt_f32_f16<<<(int)((nw / 4 + T - 1) / T), T, 0, stream>>>(Wk, Wkh, (int)(nw / 4));
  cvt_f32_f16<<<(int)((nw / 4 + T - 1) / T), T, 0, stream>>>(Wv, Wvh, (int)(nw / 4));
  cvt_f32_f16<<<(int)((nw / 4 + T - 1) / T), T, 0, stream>>>(Wo, Woh, (int)(nw / 4));

  // 32x64 tiles: (BS/32)*(1024/64) wave-tiles, 8 waves per 256-thread block
  const int gemm_tiles  = (BS / 32) * (D_MODEL / 64);
  const int gemm_blocks = (gemm_tiles + 7) / 8;
  gemm_xwt<true><<<gemm_blocks, T, 0, stream>>>(xh, Wqh, (void*)Qh,  BS, D_MODEL, D_MODEL);
  gemm_xwt<true><<<gemm_blocks, T, 0, stream>>>(xh, Wkh, (void*)Khd, BS, D_MODEL, D_MODEL);
  gemm_xwt<true><<<gemm_blocks, T, 0, stream>>>(xh, Wvh, (void*)Vh,  BS, D_MODEL, D_MODEL);

  transpose_v<<<Bsz * NUM_HEADS * (SEQ / 64), T, 0, stream>>>(Vh, Vtp);

  const int attn_waves  = Bsz * NUM_HEADS * (SEQ / 16);
  const int attn_blocks = (attn_waves + 7) / 8;
  attn_band<<<attn_blocks, T, 0, stream>>>(Qh, Khd, Vtp, Ah, Bsz);

  gemm_xwt<false><<<gemm_blocks, T, 0, stream>>>(Ah, Woh, d_out, BS, D_MODEL, D_MODEL);
}
